// MIXNet_40080634806460
// MI455X (gfx1250) — matrix operand order repaired and checked by reading, not compiled, not run on hardware
//
#include <hip/hip_runtime.h>
#include <hip/hip_bf16.h>
#include <cmath>

typedef __attribute__((ext_vector_type(16))) _Float16 v16h;
typedef __attribute__((ext_vector_type(8)))  float    v8f;

#define NUM_AGENTS 8
#define S_DIM      512
#define N1         160            // 4 * 40 fused first-layer outputs (10 tiles of 16)
#define N1T        10
#define K2         128            // padded 120 (= 3 * 40 relu'd activations)
#define N2         384            // 320 (w1) + 40 (w2) + 1 (b2) + pad  (24 tiles)
#define N2T        24

// ---- workspace layout (bytes) ----
#define WS_W1_OFF  0                                   // f16 [N1][S_DIM]   (160 KB)
#define WS_W2_OFF  (N1 * S_DIM * 2)                    // f16 [N2][K2]      (96 KB)
#define WS_B1_OFF  (WS_W2_OFF + N2 * K2 * 2)           // f32 [N1]
#define WS_B2_OFF  (WS_B1_OFF + N1 * 4)                // f32 [N2]
#define WS_G_OFF   524288                              // f32 [B][N1]       (42 MB)

#define LDS1_BYTES (N1 * S_DIM * 2)                    // 163840
#define LDS2_W2    (N2 * K2 * 2)                       // 98304
#define LDS2_BYTES (LDS2_W2 + 4 * 16 * N2 * 4)         // + plds[4][16][384] = 196608

// ---------------------------------------------------------------------------
// Kernel 0: pack weights into f16, transposed [N][K], plus fused bias vectors.
// ---------------------------------------------------------------------------
__global__ void qmix_pack(const float* __restrict__ hw1_w0, const float* __restrict__ hw2_w0,
                          const float* __restrict__ hb1_w,  const float* __restrict__ hb2_w0,
                          const float* __restrict__ hw1_b0, const float* __restrict__ hw2_b0,
                          const float* __restrict__ hb1_b,  const float* __restrict__ hb2_b0,
                          const float* __restrict__ hw1_w1, const float* __restrict__ hw1_b1,
                          const float* __restrict__ hw2_w1, const float* __restrict__ hw2_b1,
                          const float* __restrict__ hb2_w1, const float* __restrict__ hb2_b1,
                          char* __restrict__ ws) {
    _Float16* W1 = (_Float16*)(ws + WS_W1_OFF);
    _Float16* W2 = (_Float16*)(ws + WS_W2_OFF);
    float*    B1 = (float*)(ws + WS_B1_OFF);
    float*    B2 = (float*)(ws + WS_B2_OFF);
    const int tid = threadIdx.x;

    // W1T[n][k]: n 0-39 -> hw1_w0, 40-79 -> hw2_w0, 80-119 -> hb1_w, 120-159 -> hb2_w0
    for (int i = tid; i < N1 * S_DIM; i += blockDim.x) {
        int n = i >> 9, k = i & (S_DIM - 1);
        const float* src; int nn;
        if (n < 40)       { src = hw1_w0; nn = n; }
        else if (n < 80)  { src = hw2_w0; nn = n - 40; }
        else if (n < 120) { src = hb1_w;  nn = n - 80; }
        else              { src = hb2_w0; nn = n - 120; }
        W1[i] = (_Float16)src[k * 40 + nn];
    }
    for (int i = tid; i < N1; i += blockDim.x) {
        const float* src; int nn;
        if (i < 40)       { src = hw1_b0; nn = i; }
        else if (i < 80)  { src = hw2_b0; nn = i - 40; }
        else if (i < 120) { src = hb1_b;  nn = i - 80; }
        else              { src = hb2_b0; nn = i - 120; }
        B1[i] = src[nn];
    }
    // W2T[n][k]: block-diagonal second layer, K = [relu(g1)|relu(g2)|relu(g3)|pad]
    for (int i = tid; i < N2 * K2; i += blockDim.x) {
        int n = i >> 7, k = i & (K2 - 1);
        float v = 0.0f;
        if (n < 320)      { if (k < 40)             v = hw1_w1[k * 320 + n]; }
        else if (n < 360) { if (k >= 40 && k < 80)  v = hw2_w1[(k - 40) * 40 + (n - 320)]; }
        else if (n == 360){ if (k >= 80 && k < 120) v = hb2_w1[k - 80]; }
        W2[i] = (_Float16)v;
    }
    for (int i = tid; i < N2; i += blockDim.x) {
        float v = 0.0f;
        if (i < 320)       v = hw1_b1[i];
        else if (i < 360)  v = hw2_b1[i - 320];
        else if (i == 360) v = hb2_b1[0];
        B2[i] = v;
    }
}

// ---------------------------------------------------------------------------
// Kernel 1: G[B,160] = states[B,512] @ W1T' + bias1   (f16 WMMA, fp32 acc)
// 256 threads = 8 waves; W1 staged in 160 KB LDS once per 128-row block.
// ---------------------------------------------------------------------------
__global__ __launch_bounds__(256) void qmix_stage1(const float* __restrict__ states,
                                                   const char* __restrict__ ws,
                                                   float* __restrict__ G) {
    extern __shared__ char smem1[];
    _Float16* w1lds = (_Float16*)smem1;
    const float* bias1 = (const float*)(ws + WS_B1_OFF);

    {   // stage 160 KB of packed weights: 10240 uint4, 40 per thread
        const uint4* src = (const uint4*)(ws + WS_W1_OFF);
        uint4* dst = (uint4*)w1lds;
        for (int i = threadIdx.x; i < LDS1_BYTES / 16; i += 256) dst[i] = src[i];
    }
    __syncthreads();

    const int wave = threadIdx.x >> 5;
    const int lane = threadIdx.x & 31;
    const int ln   = lane & 15;          // row within tile (A) / column (B,C)
    const int hi   = lane >> 4;          // K half selector
    const int mbase = blockIdx.x * 128 + wave * 16;
    const int row   = mbase + ln;
    const float* srow = states + (size_t)row * S_DIM + hi * 16;

    v8f acc[N1T];
    const v8f vzero = {0.f, 0.f, 0.f, 0.f, 0.f, 0.f, 0.f, 0.f};
#pragma unroll
    for (int t = 0; t < N1T; ++t) acc[t] = vzero;

    for (int s = 0; s < S_DIM / 32; ++s) {
        const float* sp = srow + s * 32;
        __builtin_prefetch(sp + 32, 0, 1);                 // global_prefetch_b8
        float4 x0 = *(const float4*)(sp);
        float4 x1 = *(const float4*)(sp + 4);
        float4 x2 = *(const float4*)(sp + 8);
        float4 x3 = *(const float4*)(sp + 12);
        v16h a;
        a[0]=(_Float16)x0.x; a[1]=(_Float16)x0.y; a[2]=(_Float16)x0.z; a[3]=(_Float16)x0.w;
        a[4]=(_Float16)x1.x; a[5]=(_Float16)x1.y; a[6]=(_Float16)x1.z; a[7]=(_Float16)x1.w;
        a[8]=(_Float16)x2.x; a[9]=(_Float16)x2.y; a[10]=(_Float16)x2.z; a[11]=(_Float16)x2.w;
        a[12]=(_Float16)x3.x; a[13]=(_Float16)x3.y; a[14]=(_Float16)x3.z; a[15]=(_Float16)x3.w;

        const int kb = s * 32 + hi * 16;
        // grouped B loads from LDS so ds_loads clause ahead of the WMMA batch
#pragma unroll
        for (int g = 0; g < 2; ++g) {
            v16h b[5];
#pragma unroll
            for (int t = 0; t < 5; ++t)
                b[t] = *(const v16h*)(&w1lds[(size_t)((g * 5 + t) * 16 + ln) * S_DIM + kb]);
#pragma unroll
            for (int t = 0; t < 5; ++t)
                acc[g * 5 + t] = __builtin_amdgcn_wmma_f32_16x16x32_f16(
                    false, a, false, b[t], (short)0, acc[g * 5 + t], false, false);
        }
    }
    // C layout: lane (hi,ln) -> col = t*16+ln ; VGPR r -> row mbase + r + 8*hi
#pragma unroll
    for (int t = 0; t < N1T; ++t) {
        const int col = t * 16 + ln;
        const float bv = bias1[col];
#pragma unroll
        for (int r = 0; r < 8; ++r) {
            const int m = mbase + r + hi * 8;
            G[(size_t)m * N1 + col] = acc[t][r] + bv;
        }
    }
}

// 8 consecutive activations of X2 for one row, starting at padded-K index k0.
struct f8 { float v[8]; };
__device__ __forceinline__ f8 load_act8(const float* __restrict__ grow, int k0) {
    f8 r;
    if (k0 >= 120) {
#pragma unroll
        for (int j = 0; j < 8; ++j) r.v[j] = 0.0f;
        return r;
    }
    const int c0 = (k0 < 80) ? k0 : k0 + 40;   // skip b1 columns 80..119
    float4 x0 = *(const float4*)(grow + c0);
    float4 x1 = *(const float4*)(grow + c0 + 4);
    r.v[0] = fmaxf(x0.x, 0.f); r.v[1] = fmaxf(x0.y, 0.f);
    r.v[2] = fmaxf(x0.z, 0.f); r.v[3] = fmaxf(x0.w, 0.f);
    r.v[4] = fmaxf(x1.x, 0.f); r.v[5] = fmaxf(x1.y, 0.f);
    r.v[6] = fmaxf(x1.z, 0.f); r.v[7] = fmaxf(x1.w, 0.f);
    return r;
}

// ---------------------------------------------------------------------------
// Kernel 2: P[B,384] = relu-blocks(G) @ W2T' + bias2, abs; then per-row mix.
// 256 threads = 8 waves = 4 wave-pairs; each pair owns a 16-row tile and the
// two waves split the 24 N-tiles (12 each). Block loops over 4x64 rows so the
// 96 KB W2 is staged in LDS once per 256 rows.
// ---------------------------------------------------------------------------
__global__ __launch_bounds__(256) void qmix_stage2(const float* __restrict__ G,
                                                   const float* __restrict__ qvals,
                                                   const char* __restrict__ ws,
                                                   float* __restrict__ out) {
    extern __shared__ char smem2[];
    _Float16* w2lds = (_Float16*)smem2;                      // [N2][K2] f16
    float*    plds  = (float*)(smem2 + LDS2_W2);             // [4][16][N2] f32
    const float* bias2 = (const float*)(ws + WS_B2_OFF);

    {   // stage 96 KB of packed second-layer weights
        const uint4* src = (const uint4*)(ws + WS_W2_OFF);
        uint4* dst = (uint4*)w2lds;
        for (int i = threadIdx.x; i < LDS2_W2 / 16; i += 256) dst[i] = src[i];
    }
    __syncthreads();

    const int wave = threadIdx.x >> 5;
    const int lane = threadIdx.x & 31;
    const int ln   = lane & 15;
    const int hi   = lane >> 4;
    const int wpair = wave >> 1;         // 0..3 : which 16-row tile
    const int whalf = wave & 1;          // 0/1  : which 12 N-tiles
    const int t0    = whalf * 12;

    const v8f vzero = {0.f, 0.f, 0.f, 0.f, 0.f, 0.f, 0.f, 0.f};

    for (int iter = 0; iter < 4; ++iter) {
        const int mbase = blockIdx.x * 256 + iter * 64 + wpair * 16;
        const int row   = mbase + ln;
        const float* grow = G + (size_t)row * N1;

        v8f acc[12];
#pragma unroll
        for (int t = 0; t < 12; ++t) acc[t] = vzero;

#pragma unroll
        for (int s = 0; s < K2 / 32; ++s) {
            const int k0 = s * 32 + hi * 16;
            f8 lo = load_act8(grow, k0);
            f8 hi8 = load_act8(grow, k0 + 8);
            v16h a;
#pragma unroll
            for (int j = 0; j < 8; ++j) { a[j] = (_Float16)lo.v[j]; a[j + 8] = (_Float16)hi8.v[j]; }
            const int kb = s * 32 + hi * 16;
#pragma unroll
            for (int g = 0; g < 3; ++g) {
                v16h b[4];
#pragma unroll
                for (int t = 0; t < 4; ++t)
                    b[t] = *(const v16h*)(&w2lds[(size_t)((t0 + g * 4 + t) * 16 + ln) * K2 + kb]);
#pragma unroll
                for (int t = 0; t < 4; ++t)
                    acc[g * 4 + t] = __builtin_amdgcn_wmma_f32_16x16x32_f16(
                        false, a, false, b[t], (short)0, acc[g * 4 + t], false, false);
            }
        }

        __syncthreads();   // previous iteration's epilogue has finished with plds
        // stage C tiles: abs(P + bias2) for n<360 (w1,w2), raw for n==360 (b2)
#pragma unroll
        for (int t = 0; t < 12; ++t) {
            const int n = (t0 + t) * 16 + ln;
            const float bv = bias2[n];
#pragma unroll
            for (int r = 0; r < 8; ++r) {
                const int m = r + hi * 8;
                float v = acc[t][r] + bv;
                if (n < 360) v = fabsf(v);
                plds[(size_t)(wpair * 16 + m) * N2 + n] = v;
            }
        }
        __syncthreads();

        // Per-row epilogue: the wave-pair's 16 rows, 4 lanes per row (8 rows/wave).
        const int lr  = whalf * 8 + (lane >> 2);   // 0..15 within the pair
        const int par = lane & 3;
        const int orow = mbase - ln + lr;          // mbase is ln-free already
        const float* P   = &plds[(size_t)(wpair * 16 + lr) * N2];
        const float* b1p = G + (size_t)orow * N1 + 80;
        float qv[NUM_AGENTS];
#pragma unroll
        for (int a2 = 0; a2 < NUM_AGENTS; ++a2) qv[a2] = qvals[(size_t)orow * NUM_AGENTS + a2];

        float s = 0.0f;
        for (int h = par; h < 40; h += 4) {
            float hp = b1p[h];
#pragma unroll
            for (int a2 = 0; a2 < NUM_AGENTS; ++a2) hp = fmaf(qv[a2], P[a2 * 40 + h], hp);
            const float hid = hp > 0.0f ? hp : (__expf(hp) - 1.0f);   // ELU
            s = fmaf(hid, P[320 + h], s);
        }
        s += __shfl_xor(s, 1, 32);
        s += __shfl_xor(s, 2, 32);
        if (par == 0) out[orow] = s + P[360];
    }
}

// ---------------------------------------------------------------------------
extern "C" void kernel_launch(void* const* d_in, const int* in_sizes, int n_in,
                              void* d_out, int out_size, void* d_ws, size_t ws_size,
                              hipStream_t stream) {
    (void)n_in; (void)out_size; (void)ws_size;
    const float* q      = (const float*)d_in[0];
    const float* states = (const float*)d_in[1];
    const float* hw1_w0 = (const float*)d_in[2];
    const float* hw1_b0 = (const float*)d_in[3];
    const float* hw1_w1 = (const float*)d_in[4];
    const float* hw1_b1 = (const float*)d_in[5];
    const float* hw2_w0 = (const float*)d_in[6];
    const float* hw2_b0 = (const float*)d_in[7];
    const float* hw2_w1 = (const float*)d_in[8];
    const float* hw2_b1 = (const float*)d_in[9];
    const float* hb1_w  = (const float*)d_in[10];
    const float* hb1_b  = (const float*)d_in[11];
    const float* hb2_w0 = (const float*)d_in[12];
    const float* hb2_b0 = (const float*)d_in[13];
    const float* hb2_w1 = (const float*)d_in[14];
    const float* hb2_b1 = (const float*)d_in[15];

    const int B = in_sizes[0] / NUM_AGENTS;   // 65536
    char*  ws = (char*)d_ws;
    float* G  = (float*)(ws + WS_G_OFF);

    qmix_pack<<<1, 256, 0, stream>>>(hw1_w0, hw2_w0, hb1_w, hb2_w0,
                                     hw1_b0, hw2_b0, hb1_b, hb2_b0,
                                     hw1_w1, hw1_b1, hw2_w1, hw2_b1,
                                     hb2_w1, hb2_b1, ws);
    qmix_stage1<<<B / 128, 256, LDS1_BYTES, stream>>>(states, ws, G);
    qmix_stage2<<<B / 256, 256, LDS2_BYTES, stream>>>(G, q, ws, (float*)d_out);
}
